// GeometricAttention_72078141161692
// MI455X (gfx1250) — matrix operand order repaired and checked by reading, not compiled, not run on hardware
//
#include <hip/hip_runtime.h>

typedef _Float16 half_t;
typedef __attribute__((ext_vector_type(16))) _Float16 v16h;
typedef __attribute__((ext_vector_type(8)))  _Float16 v8h;
typedef __attribute__((ext_vector_type(8)))  float    v8f;
typedef __attribute__((ext_vector_type(4)))  unsigned u32x4;
typedef __attribute__((ext_vector_type(8)))  int      i32x8;
typedef __attribute__((ext_vector_type(4)))  int      i32x4;

#define NV 4
#define NB 2
#define NN 256
#define ND 128
#define NH 4
#define HD 32
#define MV 32768          // 32^3 grid points
#define NPB 64            // points per softmax group
#define NPAIR 12
#define INV_TEMP 10.0f

#if defined(__gfx1250__) && __has_builtin(__builtin_amdgcn_tensor_load_to_lds) && \
    __has_builtin(__builtin_amdgcn_s_wait_tensorcnt)
#define HAVE_TDM 1
#else
#define HAVE_TDM 0
#endif

static __device__ __forceinline__ v8f wmma16(v16h a, v16h b, v8f c) {
  // v_wmma_f32_16x16x32_f16 (codegen-confirmed builtin, 8 args)
  return __builtin_amdgcn_wmma_f32_16x16x32_f16(
      /*neg_a=*/false, a, /*neg_b=*/false, b,
      /*c_mod=*/(short)0, c, /*reuse_a=*/false, /*reuse_b=*/false);
}

static __device__ __forceinline__ float wave_max(float v) {
#pragma unroll
  for (int o = 16; o > 0; o >>= 1) v = fmaxf(v, __shfl_xor(v, o, 32));
  return v;
}
static __device__ __forceinline__ float wave_sum(float v) {
#pragma unroll
  for (int o = 16; o > 0; o >>= 1) v += __shfl_xor(v, o, 32);
  return v;
}

// ---------------------------------------------------------------------------
// Transpose + f16-convert a 128x128 weight matrix: WhT[n*128+k] = W[k*128+n]
// ---------------------------------------------------------------------------
__global__ void __launch_bounds__(256)
transpose_w16_kernel(const float* __restrict__ W, half_t* __restrict__ WhT) {
  int i = blockIdx.x * 256 + threadIdx.x;       // i = n*128 + k
  WhT[i] = (half_t)W[(i & 127) * ND + (i >> 7)];
}

// ---------------------------------------------------------------------------
// Track encoder: feat = relu(xy @ w1 + b1) @ w2 + b2   (rows = V*B*N)
// Writes f32 row-major (fusion pass-1 distances), f16 row-major (GEMM A),
// and f16 transposed [v,b][d][n] (fusion/WMMA B operand).
// ---------------------------------------------------------------------------
__global__ void __launch_bounds__(128)
track_encoder_kernel(const float* __restrict__ tracks,
                     const float* __restrict__ w1, const float* __restrict__ b1,
                     const float* __restrict__ w2, const float* __restrict__ b2,
                     float* __restrict__ feat, half_t* __restrict__ feat16R,
                     half_t* __restrict__ feat16T) {
  int row = blockIdx.x;                         // (v*NB+b)*NN + n
  int vb = row >> 8;
  int n = row & (NN - 1);
  int t = threadIdx.x;
  float x = tracks[row * 2 + 0];
  float y = tracks[row * 2 + 1];
  float acc = b2[t];
#pragma unroll 8
  for (int h = 0; h < 64; ++h) {
    float hv = fmaxf(x * w1[h] + y * w1[64 + h] + b1[h], 0.0f);
    acc += hv * w2[h * ND + t];
  }
  feat[(size_t)row * ND + t] = acc;
  feat16R[(size_t)row * ND + t] = (half_t)acc;
  feat16T[((size_t)vb * ND + t) * NN + n] = (half_t)acc;
}

// ---------------------------------------------------------------------------
// Grid encoder: enc = relu(xyz @ ge_w1 + ge_b1) @ ge_w2 + ge_b2  (rows = M)
// ---------------------------------------------------------------------------
__global__ void __launch_bounds__(128)
grid_encoder_kernel(const float* __restrict__ w1, const float* __restrict__ b1,
                    const float* __restrict__ w2, const float* __restrict__ b2,
                    float* __restrict__ enc) {
  int m = blockIdx.x;
  int t = threadIdx.x;
  int ix = m >> 10, iy = (m >> 5) & 31, iz = m & 31;
  const float step = 2.0f / 31.0f;
  float x = -1.0f + ix * step;
  float y = -1.0f + iy * step;
  float z = -1.0f + iz * step;
  float acc = b2[t];
#pragma unroll 8
  for (int h = 0; h < 64; ++h) {
    float hv = fmaxf(x * w1[h] + y * w1[64 + h] + z * w1[128 + h] + b1[h], 0.0f);
    acc += hv * w2[h * ND + t];
  }
  enc[(size_t)m * ND + t] = acc;
}

// ---------------------------------------------------------------------------
// Perspective-project grid into each view: grid2d[v,b,m,2]
// ---------------------------------------------------------------------------
__global__ void __launch_bounds__(256)
grid2d_kernel(const float* __restrict__ proj, float* __restrict__ grid2d) {
  int idx = blockIdx.x * 256 + threadIdx.x;     // < V*B*M
  int m = idx & (MV - 1);
  int vb = idx >> 15;
  int ix = m >> 10, iy = (m >> 5) & 31, iz = m & 31;
  const float step = 2.0f / 31.0f;
  float x = -1.0f + ix * step;
  float y = -1.0f + iy * step;
  float z = -1.0f + iz * step;
  const float* P = proj + vb * 12;
  float h0 = P[0] * x + P[1] * y + P[2] * z + P[3];
  float h1 = P[4] * x + P[5] * y + P[6] * z + P[7];
  float h2 = P[8] * x + P[9] * y + P[10] * z + P[11];
  float inv = 1.0f / (h2 + 1e-10f);
  grid2d[(size_t)idx * 2 + 0] = h0 * inv;
  grid2d[(size_t)idx * 2 + 1] = h1 * inv;
}

// ---------------------------------------------------------------------------
// Generic GEMM: C[rows x 128] = A16[rows x 128] @ W[128 x 128] + bias
// A16 = row-major f16 (pre-converted by producer; no conversions in the
// WMMA loop). WhT = pre-transposed f16 weights staged to LDS via the Tensor
// Data Mover (fallback: cooperative copy). Each wave: 16 rows x 128 cols via
// 8 n-tiles x 4 K-chunks of v_wmma_f32_16x16x32_f16. All operand fetches
// are contiguous aligned vector loads.
// ---------------------------------------------------------------------------
__global__ void __launch_bounds__(128)
gemm128_kernel(const half_t* __restrict__ A16, const half_t* __restrict__ WhT,
               const float* __restrict__ bias, float* __restrict__ Cf,
               half_t* __restrict__ Ch) {
  __shared__ __align__(32) half_t wlds[ND * ND];   // 32 KB, layout [n][k]
  __shared__ float blds[ND];
  int tid = threadIdx.x;
  int wave = tid >> 5;
  int lane = tid & 31;

#if HAVE_TDM
  if (wave == 0) {
    // 1-D TDM descriptor (ISA 8.3/8.4): copy 8192 dwords (32 KB) of f16
    // weights from global to LDS.
    unsigned lds_lo = (unsigned)(size_t)(void*)&wlds[0];
    unsigned long long ga = (unsigned long long)(size_t)WhT;
    const unsigned ndw = (ND * ND) / 2;         // 8192 dwords
    u32x4 g0;
    g0[0] = 1u;                                  // count=1, user descriptor
    g0[1] = lds_lo;                              // lds_addr
    g0[2] = (unsigned)(ga & 0xffffffffu);        // global_addr[31:0]
    g0[3] = (unsigned)((ga >> 32) & 0x01ffffffu) // global_addr[56:32]
            | (2u << 30);                        // type=2 ("image")
    i32x8 g1;
    g1[0] = (int)(2u << 16);                     // workgroup_mask=0, data_size=4B
    g1[1] = (int)((ndw & 0xffffu) << 16);        // tensor_dim0[15:0]
    g1[2] = (int)(((ndw >> 16) & 0xffffu) | (1u << 16));  // td0 hi | tensor_dim1=1
    g1[3] = (int)((ndw & 0xffffu) << 16);        // td1 hi=0 | tile_dim0=8192
    g1[4] = 1;                                   // tile_dim1=1, tile_dim2=0
    g1[5] = (int)ndw;                            // tensor_dim0_stride[31:0]
    g1[6] = 0;
    g1[7] = 0;
    i32x4 gz;
    gz[0] = gz[1] = gz[2] = gz[3] = 0;
#if __clang_major__ >= 23
    i32x8 gz8;
#pragma unroll
    for (int q = 0; q < 8; ++q) gz8[q] = 0;
    __builtin_amdgcn_tensor_load_to_lds(g0, g1, gz, gz, gz8, 0);
#else
    __builtin_amdgcn_tensor_load_to_lds(g0, g1, gz, gz, 0);
#endif
    __builtin_amdgcn_s_wait_tensorcnt(0);
  }
#else
  for (int i = tid; i < ND * ND; i += 128) wlds[i] = WhT[i];
#endif
  if (tid < ND) blds[tid] = bias[tid];
  __syncthreads();

  int ml = lane & 15;        // A row within tile / C column index
  int ch = lane >> 4;        // half-wave selector
  int r0 = blockIdx.x * 64 + wave * 16;

  v8f acc[8];
#pragma unroll
  for (int nt = 0; nt < 8; ++nt) {
    float bv = blds[nt * 16 + ml];
#pragma unroll
    for (int r = 0; r < 8; ++r) acc[nt][r] = bv;
  }

  const half_t* arow = A16 + (size_t)(r0 + ml) * ND;
#pragma unroll
  for (int kc = 0; kc < 4; ++kc) {
    // A-layout: halves 0-7 -> K = 16*ch + h, halves 8-15 -> K = 16*ch + 8 + h
    v8h alo = *(const v8h*)(arow + kc * 32 + 16 * ch);
    v8h ahi = *(const v8h*)(arow + kc * 32 + 8 + 16 * ch);
    v16h a;
#pragma unroll
    for (int h = 0; h < 8; ++h) {
      a[h]     = alo[h];
      a[h + 8] = ahi[h];
    }
#pragma unroll
    for (int nt = 0; nt < 8; ++nt) {
      // B-layout: lane N = nt*16+ml, halves h -> K = kc*32 + 16*ch + h
      v16h bm = *(const v16h*)(&wlds[(nt * 16 + ml) * ND + kc * 32 + 16 * ch]);
      acc[nt] = wmma16(a, bm, acc[nt]);
    }
  }

#pragma unroll
  for (int nt = 0; nt < 8; ++nt) {
#pragma unroll
    for (int r = 0; r < 8; ++r) {
      int row = r0 + r + 8 * ch;                            // C-layout M
      int nc = nt * 16 + ml;                                // C-layout N
      float v = acc[nt][r];
      if (Cf) Cf[(size_t)row * ND + nc] = v;
      if (Ch) Ch[(size_t)row * ND + nc] = (half_t)v;
    }
  }
}

// ---------------------------------------------------------------------------
// Cross-view attention scores: one 16x16 tile per wave, K = 32 (one WMMA).
// scores[pi,b,h,n,m] = (q[i,b,n,h,:] . k[j,b,m,h,:]) / sqrt(32)
// ---------------------------------------------------------------------------
__global__ void __launch_bounds__(32)
scores_kernel(const half_t* __restrict__ q16, const half_t* __restrict__ k16,
              float* __restrict__ scores) {
  int bid = blockIdx.x;
  int mt = bid & 15; bid >>= 4;
  int nt = bid & 15; bid >>= 4;
  int hh = bid & 3;  bid >>= 2;
  int b  = bid & 1;  bid >>= 1;
  int pi = bid;                                 // 0..11
  int i = pi / 3;
  int jj = pi % 3;
  int j = jj + (jj >= i ? 1 : 0);

  int lane = threadIdx.x;
  int ml = lane & 15;
  int ch = lane >> 4;

  const half_t* qrow =
      q16 + ((size_t)(i * NB + b) * NN + nt * 16 + ml) * ND + hh * HD;
  v8h qlo = *(const v8h*)(qrow + 16 * ch);
  v8h qhi = *(const v8h*)(qrow + 8 + 16 * ch);
  v16h a;
#pragma unroll
  for (int h = 0; h < 8; ++h) {
    a[h]     = qlo[h];
    a[h + 8] = qhi[h];
  }
  const half_t* krow =
      k16 + ((size_t)(j * NB + b) * NN + mt * 16 + ml) * ND + hh * HD;
  v16h bm = *(const v16h*)(krow + 16 * ch);

  v8f c;
#pragma unroll
  for (int r = 0; r < 8; ++r) c[r] = 0.0f;
  c = wmma16(a, bm, c);

  const float scale = 0.17677669529663687f;     // 1/sqrt(32)
  float* sbase = scores + ((size_t)(pi * NB + b) * NH + hh) * NN * NN;
#pragma unroll
  for (int r = 0; r < 8; ++r) {
    int n = nt * 16 + r + 8 * ch;
    int m = mt * 16 + ml;
    sbase[(size_t)n * NN + m] = c[r] * scale;
  }
}

// ---------------------------------------------------------------------------
// Row softmax over m, mean over heads -> corr output.
// Wave32 shuffle reductions + one 8-entry LDS combine (4 barriers per head
// instead of ~18 with a shared-memory tree).
// ---------------------------------------------------------------------------
__global__ void __launch_bounds__(256)
softmax_corr_kernel(const float* __restrict__ scores, float* __restrict__ corr) {
  __shared__ float rbuf[8];
  int bid = blockIdx.x;
  int n = bid % NN;
  int pb = bid / NN;
  int b = pb % NB;
  int pi = pb / NB;
  int m = threadIdx.x;
  int wid = m >> 5;
  int ln = m & 31;

  float accw = 0.0f;
  for (int hh = 0; hh < NH; ++hh) {
    const float* row =
        scores + (((size_t)(pi * NB + b) * NH + hh) * NN + n) * NN;
    float s = row[m];
    float wm = wave_max(s);
    if (ln == 0) rbuf[wid] = wm;
    __syncthreads();
    float mx = rbuf[0];
#pragma unroll
    for (int w = 1; w < 8; ++w) mx = fmaxf(mx, rbuf[w]);
    __syncthreads();
    float e = __expf(s - mx);
    float ws = wave_sum(e);
    if (ln == 0) rbuf[wid] = ws;
    __syncthreads();
    float sm = rbuf[0];
#pragma unroll
    for (int w = 1; w < 8; ++w) sm += rbuf[w];
    __syncthreads();
    accw += e / sm;
  }
  corr[((size_t)(pi * NB + b) * NN + n) * NN + m] = accw * 0.25f;
}

// ---------------------------------------------------------------------------
// Fusion attention: one wave per (b, 16-grid-point tile).
// Lanes 0-15 own softmax groups {0,1}, lanes 16-31 own {2,3} of their row, so
// each 64-point softmax lives entirely inside one lane (no reductions).
// Pass 1 normalizes weights into an f16 LDS buffer, so pass 2's WMMA loop has
// zero conversions: A = two aligned v8h LDS loads, B = one aligned 32-byte
// global load from the transposed f16 features. Epilogue writes vol as f16
// (the final GEMM's A operand) directly.
// ---------------------------------------------------------------------------
__global__ void __launch_bounds__(32)
fusion_kernel(const float* __restrict__ feat, const half_t* __restrict__ feat16T,
              const float* __restrict__ grid2d, const float* __restrict__ enc,
              half_t* __restrict__ vol16) {
  __shared__ __align__(32) float  sbuf[16 * NN];   // 16 KB raw scores / exps
  __shared__ __align__(32) half_t wbuf[16 * NN];   // 8 KB normalized weights
  int bid = blockIdx.x;                         // b*(M/16) + mtile
  int mt = bid & 2047;
  int b = bid >> 11;
  int m0 = mt * 16;
  int lane = threadIdx.x;
  int ml = lane & 15;
  int ch = lane >> 4;

  v8f acc[8];
#pragma unroll
  for (int nt = 0; nt < 8; ++nt)
#pragma unroll
    for (int r = 0; r < 8; ++r) acc[nt][r] = 0.0f;

  for (int v = 0; v < NV; ++v) {
    float gx = grid2d[((size_t)(v * NB + b) * MV + m0 + ml) * 2 + 0];
    float gy = grid2d[((size_t)(v * NB + b) * MV + m0 + ml) * 2 + 1];
    const float* fbase = feat + (size_t)(v * NB + b) * NN * ND;
    const half_t* f16base = feat16T + (size_t)(v * NB + b) * ND * NN;
    __builtin_prefetch(f16base, 0, 1);          // global_prefetch_b8

    // pass 1: per-group streaming softmax; normalized f16 weights -> wbuf
#pragma unroll
    for (int gi = 0; gi < 2; ++gi) {
      int g = 2 * ch + gi;
      float mx = -3.0e38f;
      for (int pl = 0; pl < NPB; ++pl) {
        int p = g * NPB + pl;
        float dx = gx - fbase[(size_t)p * ND + 0];
        float dy = gy - fbase[(size_t)p * ND + 1];
        float s = -sqrtf(fmaxf(dx * dx + dy * dy, 0.0f)) * INV_TEMP;
        sbuf[ml * NN + p] = s;
        mx = fmaxf(mx, s);
      }
      float sum = 0.0f;
      for (int pl = 0; pl < NPB; ++pl) {
        int p = g * NPB + pl;
        float e = __expf(sbuf[ml * NN + p] - mx);
        sbuf[ml * NN + p] = e;
        sum += e;
      }
      float inv = 1.0f / sum;
      for (int pl = 0; pl < NPB; ++pl) {
        int p = g * NPB + pl;
        wbuf[ml * NN + p] = (half_t)(sbuf[ml * NN + p] * inv);
      }
    }
    __syncthreads();

    // pass 2: fusion += attn(16x256) @ feat(256x128) via 8x8 WMMA tiles
#pragma unroll 2
    for (int kc = 0; kc < 8; ++kc) {
      v8h wlo = *(const v8h*)(&wbuf[ml * NN + kc * 32 + 16 * ch]);
      v8h whi = *(const v8h*)(&wbuf[ml * NN + kc * 32 + 8 + 16 * ch]);
      v16h a;
#pragma unroll
      for (int h = 0; h < 8; ++h) {
        a[h]     = wlo[h];
        a[h + 8] = whi[h];
      }
#pragma unroll
      for (int nt = 0; nt < 8; ++nt) {
        v16h bm =
            *(const v16h*)(f16base + (size_t)(nt * 16 + ml) * NN + kc * 32 + 16 * ch);
        acc[nt] = wmma16(a, bm, acc[nt]);
      }
    }
    __syncthreads();
  }

#pragma unroll
  for (int nt = 0; nt < 8; ++nt) {
#pragma unroll
    for (int r = 0; r < 8; ++r) {
      int mg = m0 + r + 8 * ch;
      int nc = nt * 16 + ml;
      vol16[((size_t)b * MV + mg) * ND + nc] =
          (half_t)(enc[(size_t)mg * ND + nc] + acc[nt][r] * 0.25f);
    }
  }
}

// ---------------------------------------------------------------------------
// Host-side orchestration
// ---------------------------------------------------------------------------
extern "C" void kernel_launch(void* const* d_in, const int* in_sizes, int n_in,
                              void* d_out, int out_size, void* d_ws,
                              size_t ws_size, hipStream_t stream) {
  const float* tracks = (const float*)d_in[0];
  const float* projm  = (const float*)d_in[1];
  const float* te_w1  = (const float*)d_in[2];
  const float* te_b1  = (const float*)d_in[3];
  const float* te_w2  = (const float*)d_in[4];
  const float* te_b2  = (const float*)d_in[5];
  const float* ge_w1  = (const float*)d_in[6];
  const float* ge_b1  = (const float*)d_in[7];
  const float* ge_w2  = (const float*)d_in[8];
  const float* ge_b2  = (const float*)d_in[9];
  const float* wq     = (const float*)d_in[10];
  const float* bq     = (const float*)d_in[11];
  const float* wk     = (const float*)d_in[12];
  const float* bk     = (const float*)d_in[13];
  const float* pr_w   = (const float*)d_in[14];
  const float* pr_b   = (const float*)d_in[15];

  float* out = (float*)d_out;
  float* corr_out = out;                                  // (12,B,N,N)
  float* proj_out = out + (size_t)NPAIR * NB * NN * NN;   // (B,32,32,32,D)

  char* ws = (char*)d_ws;
  float*  feat    = (float*)ws;  ws += (size_t)NV * NB * NN * ND * 4;
  half_t* feat16R = (half_t*)ws; ws += (size_t)NV * NB * NN * ND * 2;
  half_t* feat16T = (half_t*)ws; ws += (size_t)NV * NB * ND * NN * 2;
  half_t* q16     = (half_t*)ws; ws += (size_t)NV * NB * NN * ND * 2;
  half_t* k16     = (half_t*)ws; ws += (size_t)NV * NB * NN * ND * 2;
  half_t* wq16T   = (half_t*)ws; ws += (size_t)ND * ND * 2;
  half_t* wk16T   = (half_t*)ws; ws += (size_t)ND * ND * 2;
  half_t* pw16T   = (half_t*)ws; ws += (size_t)ND * ND * 2;
  half_t* vol16   = (half_t*)ws; ws += (size_t)NB * MV * ND * 2;
  float*  g2d     = (float*)ws;  ws += (size_t)NV * NB * MV * 2 * 4;
  float*  enc     = (float*)ws;  ws += (size_t)MV * ND * 4;
  float*  scores  = (float*)ws;  ws += (size_t)NPAIR * NB * NH * NN * NN * 4;

  transpose_w16_kernel<<<64, 256, 0, stream>>>(wq, wq16T);
  transpose_w16_kernel<<<64, 256, 0, stream>>>(wk, wk16T);
  transpose_w16_kernel<<<64, 256, 0, stream>>>(pr_w, pw16T);

  track_encoder_kernel<<<NV * NB * NN, 128, 0, stream>>>(
      tracks, te_w1, te_b1, te_w2, te_b2, feat, feat16R, feat16T);
  grid_encoder_kernel<<<MV, 128, 0, stream>>>(ge_w1, ge_b1, ge_w2, ge_b2, enc);
  grid2d_kernel<<<(NV * NB * MV) / 256, 256, 0, stream>>>(projm, g2d);

  gemm128_kernel<<<(NV * NB * NN) / 64, 128, 0, stream>>>(
      feat16R, wq16T, bq, nullptr, q16);
  gemm128_kernel<<<(NV * NB * NN) / 64, 128, 0, stream>>>(
      feat16R, wk16T, bk, nullptr, k16);

  scores_kernel<<<NPAIR * NB * NH * 16 * 16, 32, 0, stream>>>(q16, k16, scores);
  softmax_corr_kernel<<<NPAIR * NB * NN, 256, 0, stream>>>(scores, corr_out);

  fusion_kernel<<<NB * (MV / 16), 32, 0, stream>>>(feat, feat16T, g2d, enc, vol16);

  gemm128_kernel<<<(NB * MV) / 64, 128, 0, stream>>>(
      vol16, pw16T, pr_b, proj_out, nullptr);
}
